// TIG_Encoder_3813930959233
// MI455X (gfx1250) — matrix operand order repaired and checked by reading, not compiled, not run on hardware
//
#include <hip/hip_runtime.h>
#include <hip/hip_bf16.h>
#include <hip/hip_fp16.h>

// GCN 2-layer forward for MI455X (gfx1250, wave32).
// GEMMs use v_wmma_f32_16x16x32_f16 (f16 A/B converted from fp32, fp32
// accumulate). Edge aggregation uses hardware fp32 global atomics; feature
// matrices (102.4MB) fit in the 192MB L2, so gather/scatter stays on-chip.
// Workspace requirement: ~257 MB.

#define IN_C 256

typedef __attribute__((ext_vector_type(16))) _Float16 v16h;
typedef __attribute__((ext_vector_type(8)))  _Float16 v8h;
typedef __attribute__((ext_vector_type(4)))  _Float16 v4h;
typedef __attribute__((ext_vector_type(8)))  float    v8f;

// ---------------------------------------------------------------------------
// degree / normalization
// ---------------------------------------------------------------------------
__global__ void k_init_deg(float* __restrict__ deg, int n) {
    int i = blockIdx.x * blockDim.x + threadIdx.x;
    if (i < n) deg[i] = 1.0f;              // self-loop contribution
}

__global__ void k_deg_acc(const int* __restrict__ ei, float* __restrict__ deg,
                          int nedges) {
    int e = blockIdx.x * blockDim.x + threadIdx.x;
    if (e >= nedges) return;
    int d = ei[nedges + e];                // dst row of edge_index
    unsafeAtomicAdd(&deg[d], 1.0f);        // global_atomic_add_f32 (no CAS loop)
}

__global__ void k_deg_to_dis(float* __restrict__ deg, int n) {
    int i = blockIdx.x * blockDim.x + threadIdx.x;
    if (i >= n) return;
    float d = deg[i];
    deg[i] = (d > 0.0f) ? __frsqrt_rn(fmaxf(d, 1e-12f)) : 0.0f;
}

// ---------------------------------------------------------------------------
// fp32 -> f16 conversion helpers
// ---------------------------------------------------------------------------
__global__ void k_cvt_f16(const float* __restrict__ src, _Float16* __restrict__ dst,
                          int n4) {
    int i = blockIdx.x * blockDim.x + threadIdx.x;
    if (i >= n4) return;
    float4 v = ((const float4*)src)[i];
    v4h o = {(_Float16)v.x, (_Float16)v.y, (_Float16)v.z, (_Float16)v.w};
    ((v4h*)dst)[i] = o;
}

// W (K x N, row major) -> WT (N x K) in f16: makes the B tile K-contiguous.
__global__ void k_cvt_transpose(const float* __restrict__ W, _Float16* __restrict__ WT) {
    int i = blockIdx.x * blockDim.x + threadIdx.x;   // 256*256 elements
    int r = i >> 8, c = i & 255;
    WT[c * 256 + r] = (_Float16)W[i];
}

// ---------------------------------------------------------------------------
// WMMA GEMM: Y(nrows x 256, f32) = Xh(nrows x 256, f16) @ W(256 x 256)
// WT is W transposed ([col][k], f16). 256 threads = 8 waves; each wave owns a
// 16-row x 256-col strip (16 accumulators of 16x16). W staged via LDS in
// K=64 chunks, double-buffered so chunk k+1 staging overlaps chunk k WMMAs.
// LDS row stride 72 halves (144B = 9*16B): b128 accesses stay 16B-aligned and
// the per-row 36-dword bank rotation makes the 16 per-lane b128 B-fragment
// reads land on 16 distinct 4-bank groups (conflict-free).
// ---------------------------------------------------------------------------
#define KCHUNK   64
#define LDS_STR  72   // 64 + 8 halves padding
#define NCHUNKS  (IN_C / KCHUNK)

__global__ __launch_bounds__(256) void
k_gemm_wmma(const _Float16* __restrict__ Xh, const _Float16* __restrict__ WT,
            float* __restrict__ Y, int nrows) {
    __shared__ _Float16 lds[2][256 * LDS_STR];   // 2 x 36,864 bytes

    const int tid  = threadIdx.x;
    const int lane = tid & 31;
    const int wave = tid >> 5;
    const int waveRow = blockIdx.x * 128 + wave * 16;

    v8f acc[16];
#pragma unroll
    for (int n = 0; n < 16; ++n) acc[n] = (v8f)0.0f;

    const int m     = lane & 15;          // A row within tile
    const int khalf = (lane >> 4) * 8;    // A K sub-offset (lanes 16-31 hold K+8)
    const int cb    = lane & 15;          // B column within tile
    const int koff2 = (lane >> 4) * 8;    // B K sub-offset (lanes 16-31 hold K+8)
    const int arow  = waveRow + m;

    // Vectorized staging: 16384 halves / chunk = 2048 v8h; 8 per thread.
    const int sc = tid >> 3;              // WT row (output column), 0..31 step
    const int sj = (tid & 7) * 8;         // K sub-offset within chunk
    auto stage = [&](int buf, int kbase) {
#pragma unroll
        for (int it = 0; it < 8; ++it) {
            int c = sc + it * 32;
            v8h v = *(const v8h*)(WT + c * 256 + kbase + sj);
            *(v8h*)(&lds[buf][c * LDS_STR + sj]) = v;
        }
    };

    stage(0, 0);

    for (int kc = 0; kc < NCHUNKS; ++kc) {
        const int kbase = kc * KCHUNK;
        const int buf   = kc & 1;
        __syncthreads();                  // staging(kc) + compute(kc-1) done
        if (kc + 1 < NCHUNKS) stage(buf ^ 1, kbase + KCHUNK);

#pragma unroll
        for (int ks = 0; ks < KCHUNK / 32; ++ks) {
            const int kk = ks * 32;
            // ---- A fragment: 16x32 f16 (ISA 7.12.2 layout) ----
            v16h a = {};
            if (arow < nrows) {
                const _Float16* xp = Xh + (size_t)arow * IN_C + kbase + kk + khalf;
                v8h alo = *(const v8h*)xp;          // K = kk+khalf   .. +7
                v8h ahi = *(const v8h*)(xp + 16);   // K = kk+16+khalf.. +7
                a = __builtin_shufflevector(alo, ahi, 0, 1, 2, 3, 4, 5, 6, 7,
                                            8, 9, 10, 11, 12, 13, 14, 15);
            }
            // ---- 16 B fragments (32x16 f16 each) + WMMA ----
#pragma unroll
            for (int n = 0; n < 16; ++n) {
                const _Float16* bp = &lds[buf][(n * 16 + cb) * LDS_STR + kk + koff2];
                v8h blo = *(const v8h*)bp;          // K = kk+koff2    .. +7
                v8h bhi = *(const v8h*)(bp + 16);   // K = kk+16+koff2 .. +7
                v16h b = __builtin_shufflevector(blo, bhi, 0, 1, 2, 3, 4, 5, 6, 7,
                                                 8, 9, 10, 11, 12, 13, 14, 15);
                acc[n] = __builtin_amdgcn_wmma_f32_16x16x32_f16(
                    /*neg_a=*/false, a, /*neg_b=*/false, b,
                    /*c_mod=*/(short)0, acc[n],
                    /*reuse_a=*/false, /*reuse_b=*/false);
            }
        }
    }

    // ---- Store C/D: VGPR j -> row (j | j+8), lanes 0-15 / 16-31 ----
    const int rsel = (lane >> 4) * 8;
#pragma unroll
    for (int n = 0; n < 16; ++n) {
        const int col = n * 16 + cb;
#pragma unroll
        for (int j = 0; j < 8; ++j) {
            int row = waveRow + rsel + j;
            if (row < nrows) Y[(size_t)row * IN_C + col] = acc[n][j];
        }
    }
}

// ---------------------------------------------------------------------------
// Aggregation
// ---------------------------------------------------------------------------
// out[i][:] = dis[i]^2 * h[i][:] + b[:]   (self-loop term + bias; also fully
// initializes the destination so the later atomics have a clean base).
__global__ void k_agg_init(const float* __restrict__ H, const float* __restrict__ dis,
                           const float* __restrict__ bias, float* __restrict__ out,
                           int n4tot) {
    int i = blockIdx.x * blockDim.x + threadIdx.x;   // over N*64 float4s
    if (i >= n4tot) return;
    int node = i >> 6;
    int c4   = i & 63;
    float ds = dis[node];
    float nl = ds * ds;
    float4 h = ((const float4*)H)[i];
    float4 b = ((const float4*)bias)[c4];
    float4 o = {nl * h.x + b.x, nl * h.y + b.y, nl * h.z + b.z, nl * h.w + b.w};
    ((float4*)out)[i] = o;
}

// One wave32 per edge: gather 1KB row of H[src] (coalesced), scale by
// dis[src]*dis[dst], fp32 atomic scatter into out[dst].
__global__ __launch_bounds__(256) void
k_scatter(const float* __restrict__ H, const int* __restrict__ ei,
          const float* __restrict__ dis, float* __restrict__ out, int nedges) {
    const int lane = threadIdx.x & 31;
    const int e = blockIdx.x * (blockDim.x >> 5) + (threadIdx.x >> 5);
    if (e >= nedges) return;
    const int s = ei[e];
    const int d = ei[nedges + e];
    const float nrm = dis[s] * dis[d];
    const float4* hs = (const float4*)(H + (size_t)s * IN_C);
    float* od = out + (size_t)d * IN_C;
#pragma unroll
    for (int k = 0; k < 2; ++k) {
        int c4 = lane + 32 * k;           // float4 index: coalesced 512B run
        float4 v = hs[c4];
        int c = c4 * 4;
        unsafeAtomicAdd(od + c + 0, nrm * v.x);
        unsafeAtomicAdd(od + c + 1, nrm * v.y);
        unsafeAtomicAdd(od + c + 2, nrm * v.z);
        unsafeAtomicAdd(od + c + 3, nrm * v.w);
    }
}

__global__ void k_prelu(float* __restrict__ H, const float* __restrict__ alpha_p,
                        int n4tot) {
    int i = blockIdx.x * blockDim.x + threadIdx.x;
    if (i >= n4tot) return;
    float a = alpha_p[0];
    float4 v = ((float4*)H)[i];
    v.x = v.x >= 0.0f ? v.x : a * v.x;
    v.y = v.y >= 0.0f ? v.y : a * v.y;
    v.z = v.z >= 0.0f ? v.z : a * v.z;
    v.w = v.w >= 0.0f ? v.w : a * v.w;
    ((float4*)H)[i] = v;
}

// ---------------------------------------------------------------------------
extern "C" void kernel_launch(void* const* d_in, const int* in_sizes, int n_in,
                              void* d_out, int out_size, void* d_ws, size_t ws_size,
                              hipStream_t stream) {
    const float* x     = (const float*)d_in[0];
    const int*   ei    = (const int*)d_in[1];     // (2, E) int32
    const float* W1    = (const float*)d_in[2];
    const float* b1    = (const float*)d_in[3];
    const float* W2    = (const float*)d_in[4];
    const float* b2    = (const float*)d_in[5];
    const float* alpha = (const float*)d_in[6];
    float* out = (float*)d_out;

    const int N = in_sizes[0] / IN_C;       // 100000
    const int E = in_sizes[1] / 2;          // 3200000

    // ---- workspace carve-up (256B aligned regions) ----
    char* base = (char*)d_ws;
    size_t off = 0;
    auto carve = [&](size_t bytes) {
        void* p = base + off;
        off = (off + bytes + 255) & ~(size_t)255;
        return p;
    };
    float*    dis = (float*)carve((size_t)N * 4);              // deg -> dis
    float*    B0  = (float*)carve((size_t)N * IN_C * 4);       // GEMM out
    float*    B1  = (float*)carve((size_t)N * IN_C * 4);       // aggregate
    _Float16* Xh  = (_Float16*)carve((size_t)N * IN_C * 2);    // f16 activations
    _Float16* W1t = (_Float16*)carve((size_t)IN_C * IN_C * 2);
    _Float16* W2t = (_Float16*)carve((size_t)IN_C * IN_C * 2);
    (void)ws_size; (void)n_in; (void)out_size;

    const int n4    = N * (IN_C / 4);                 // float4 count per matrix
    const int gN    = (N + 255) / 256;
    const int gE    = (E + 255) / 256;
    const int g4    = (n4 + 255) / 256;
    const int gGemm = (N + 127) / 128;
    const int gScat = (E + 7) / 8;                    // 8 waves per block

    // degree + symmetric normalization
    k_init_deg<<<gN, 256, 0, stream>>>(dis, N);
    k_deg_acc<<<gE, 256, 0, stream>>>(ei, dis, E);
    k_deg_to_dis<<<gN, 256, 0, stream>>>(dis, N);

    // precision conversion
    k_cvt_transpose<<<IN_C, 256, 0, stream>>>(W1, W1t);
    k_cvt_transpose<<<IN_C, 256, 0, stream>>>(W2, W2t);
    k_cvt_f16<<<g4, 256, 0, stream>>>(x, Xh, n4);

    // ---- layer 1: linear -> aggregate -> PReLU ----
    k_gemm_wmma<<<gGemm, 256, 0, stream>>>(Xh, W1t, B0, N);
    k_agg_init<<<g4, 256, 0, stream>>>(B0, dis, b1, B1, n4);
    k_scatter<<<gScat, 256, 0, stream>>>(B0, ei, dis, B1, E);
    k_prelu<<<g4, 256, 0, stream>>>(B1, alpha, n4);

    // ---- layer 2: linear -> aggregate (no activation) ----
    k_cvt_f16<<<g4, 256, 0, stream>>>(B1, Xh, n4);
    k_gemm_wmma<<<gGemm, 256, 0, stream>>>(Xh, W2t, B0, N);
    k_agg_init<<<g4, 256, 0, stream>>>(B0, dis, b2, out, n4);
    k_scatter<<<gScat, 256, 0, stream>>>(B0, ei, dis, out, E);
}
